// CircleLoss_myimplementation_79980880986260
// MI455X (gfx1250) — matrix operand order repaired and checked by reading, not compiled, not run on hardware
//
#include <hip/hip_runtime.h>
#include <math.h>

// ---------------------------------------------------------------------------
// CircleLoss forward, fused:  loss = log(1 + sum(sn) * sum(sp))
//   ff = L2-normalize(emb) rows (pre-converted to bf16 in workspace)
//   wf = ff @ W^T (256 x 100000) via v_wmma_f32_16x16x32_bf16, f32 accumulate
// Double-buffered LDS pipeline (single compute site, pointer-swapped buffers):
//   A tile  : bf16 copy  -> global_load_async_to_lds_b128 (ASYNCcnt)
//   B tile  : f32 stream -> non-temporal b128 loads -> v_cvt_pk_bf16_f32 -> ds
// ---------------------------------------------------------------------------

typedef __bf16  bf16_t;
typedef bf16_t  v8bf  __attribute__((ext_vector_type(8)));
typedef bf16_t  v16bf __attribute__((ext_vector_type(16)));
typedef float   v8f   __attribute__((ext_vector_type(8)));
typedef float   v4f   __attribute__((ext_vector_type(4)));

#define THREADS 512       // 16 wave32 waves
#define MROWS   256       // full batch per block -> W read from HBM exactly once
#define NCOLS   128       // classes per block
#define KC      32        // K chunk == WMMA K
#define KCP     40        // padded LDS row stride (bf16): 80B = 16B-aligned,
                          // 20-dword stride -> conflict-free 16-lane b128 reads

#define S_GAMMA 64.0f
#define OPT_P   1.25f
#define DELTA_P 0.75f
#define MARG    0.25f

__global__ void init_accum_kernel(double* accum) {
  accum[0] = 0.0;   // sum(sn)
  accum[1] = 0.0;   // sum(sp)
}

// One block per row: ffb[r] = bf16( emb[r] / max(||emb[r]||, 1e-12) )
__global__ void normalize_rows_kernel(const float* __restrict__ emb,
                                      bf16_t* __restrict__ ffb, int Dn) {
  __shared__ float red[256];
  const int r = blockIdx.x;
  const float* row = emb + (size_t)r * Dn;
  float s = 0.f;
  for (int d = threadIdx.x; d < Dn; d += blockDim.x) { float v = row[d]; s += v * v; }
  red[threadIdx.x] = s;
  __syncthreads();
  for (int off = 128; off > 0; off >>= 1) {
    if (threadIdx.x < off) red[threadIdx.x] += red[threadIdx.x + off];
    __syncthreads();
  }
  const float scale = 1.0f / fmaxf(sqrtf(red[0]), 1e-12f);
  bf16_t* frow = ffb + (size_t)r * Dn;
  for (int d = threadIdx.x; d < Dn; d += blockDim.x)
    frow[d] = (bf16_t)(row[d] * scale);
}

// Block tile: 256 rows x 128 classes; wave (wm,wn) owns rows[wm*64,+64) x cols[wn*32,+32)
// as a 4x2 grid of 16x16 WMMA tiles.
__global__ __launch_bounds__(THREADS)
void circle_fused_kernel(const bf16_t* __restrict__ ffb, const float* __restrict__ W,
                         const int* __restrict__ labels, double* __restrict__ accum,
                         int Dn, int Cn) {
  __shared__ bf16_t sA[2][MROWS * KCP];
  __shared__ bf16_t sB[2][NCOLS * KCP];
  __shared__ int slab[MROWS];

  const int tid   = threadIdx.x;
  const int cbase = blockIdx.x * NCOLS;
  if (tid < MROWS) slab[tid] = labels[tid];

  const int w    = tid >> 5;
  const int lane = tid & 31;
  const int wm   = w & 3;
  const int wn   = w >> 2;
  const int lm   = lane & 15;
  const int hh   = lane >> 4;

  v8f acc[4][2] = {};

  // ---- per-thread staging geometry ----
  // B: 128 rows x 8 float4 per chunk = 1024 units, 2 per thread
  const int bu0 = tid, bu1 = tid + THREADS;
  const int brow0 = bu0 >> 3, bc0 = (bu0 & 7) << 2;
  const int brow1 = bu1 >> 3, bc1 = (bu1 & 7) << 2;
  const bool bv0 = (cbase + brow0) < Cn;
  const bool bv1 = (cbase + brow1) < Cn;
  const float* wp0 = W + (size_t)(cbase + brow0) * Dn + bc0;
  const float* wp1 = W + (size_t)(cbase + brow1) * Dn + bc1;
  // A: 256 rows x 4 16B-units per chunk = 1024 units, 2 per thread
  const int au0 = tid, au1 = tid + THREADS;
  const int ar0 = au0 >> 2, ac0 = (au0 & 3) << 3;   // bf16 elem offset in row
  const int ar1 = au1 >> 2, ac1 = (au1 & 3) << 3;
  const bf16_t* apg0 = ffb + (size_t)ar0 * Dn + ac0;
  const bf16_t* apg1 = ffb + (size_t)ar1 * Dn + ac1;

  v4f wreg0, wreg1;

  // issue next chunk's global traffic: A via async copy to LDS, W into regs
  auto issue_next = [&](bf16_t* sAbuf, int k0) {
    unsigned d0 = (unsigned)(size_t)(sAbuf + ar0 * KCP + ac0);
    unsigned d1 = (unsigned)(size_t)(sAbuf + ar1 * KCP + ac1);
    asm volatile("global_load_async_to_lds_b128 %0, %1, off"
                 :: "v"(d0), "v"(apg0 + k0) : "memory");
    asm volatile("global_load_async_to_lds_b128 %0, %1, off"
                 :: "v"(d1), "v"(apg1 + k0) : "memory");
    v4f z = {0.f, 0.f, 0.f, 0.f};
    wreg0 = z; wreg1 = z;
    if (bv0) wreg0 = __builtin_nontemporal_load((const v4f*)(wp0 + k0));
    if (bv1) wreg1 = __builtin_nontemporal_load((const v4f*)(wp1 + k0));
  };

  auto store_W = [&](bf16_t* sBbuf) {
    bf16_t* d0 = sBbuf + brow0 * KCP + bc0;
    d0[0] = (bf16_t)wreg0.x; d0[1] = (bf16_t)wreg0.y;
    d0[2] = (bf16_t)wreg0.z; d0[3] = (bf16_t)wreg0.w;
    bf16_t* d1 = sBbuf + brow1 * KCP + bc1;
    d1[0] = (bf16_t)wreg1.x; d1[1] = (bf16_t)wreg1.y;
    d1[2] = (bf16_t)wreg1.z; d1[3] = (bf16_t)wreg1.w;
  };

  auto compute = [&](const bf16_t* sAbuf, const bf16_t* sBbuf) {
    // B 32x16 frag: lane half hh holds K=[8hh,+8) then [16+8hh,+8), col = lm
    v16bf bfrag[2];
#pragma unroll
    for (int j = 0; j < 2; j++) {
      int n = (wn * 2 + j) * 16 + lm;
      const bf16_t* p = sBbuf + n * KCP;
      bfrag[j] = __builtin_shufflevector(*(const v8bf*)(p + hh * 8), *(const v8bf*)(p + 16 + hh * 8),
                   0,1,2,3,4,5,6,7,8,9,10,11,12,13,14,15);
    }
    // A 16x32 frag: lane half hh holds K=[8hh,+8) then [16+8hh,+8), row = lm
#pragma unroll
    for (int i = 0; i < 4; i++) {
      int m = (wm * 4 + i) * 16 + lm;
      v8bf lo = *(const v8bf*)(sAbuf + m * KCP + hh * 8);
      v8bf hi = *(const v8bf*)(sAbuf + m * KCP + 16 + hh * 8);
      v16bf afrag = __builtin_shufflevector(lo, hi,
                      0,1,2,3,4,5,6,7,8,9,10,11,12,13,14,15);
#pragma unroll
      for (int j = 0; j < 2; j++)
        acc[i][j] = __builtin_amdgcn_wmma_f32_16x16x32_bf16(
            false, afrag, false, bfrag[j], (short)0, acc[i][j], false, false);
    }
  };

  // ---- software pipeline: single compute site, pointer-swapped buffers ----
  const int nchunks = Dn / KC;            // 16 for D=512
  bf16_t* curA = sA[0]; bf16_t* curB = sB[0];
  bf16_t* nxtA = sA[1]; bf16_t* nxtB = sB[1];

  issue_next(curA, 0);
  store_W(curB);
  asm volatile("s_wait_asynccnt 0" ::: "memory");
  __syncthreads();

  for (int ch = 0; ch < nchunks; ch++) {
    const bool more = (ch + 1) < nchunks;
    if (more) issue_next(nxtA, (ch + 1) * KC);   // overlap next chunk's HBM stream
    compute(curA, curB);                          // WMMA on current chunk
    if (more) store_W(nxtB);                      // convert + park next W tile
    asm volatile("s_wait_asynccnt 0" ::: "memory");
    __syncthreads();
    bf16_t* t;
    t = curA; curA = nxtA; nxtA = t;
    t = curB; curB = nxtB; nxtB = t;
  }

  // ---- epilogue: CircleLoss terms in-register ----
  // C/D layout: (lane, vgpr v) = (row m = 8*hh + v, col n = lm)
  float sn_p = 0.f, sp_p = 0.f;
#pragma unroll
  for (int i = 0; i < 4; i++) {
#pragma unroll
    for (int j = 0; j < 2; j++) {
      int col = cbase + (wn * 2 + j) * 16 + lm;
      if (col < Cn) {
#pragma unroll
        for (int v = 0; v < 8; v++) {
          int   row = wm * 64 + i * 16 + hh * 8 + v;
          float val = acc[i][j][v];
          if (col == slab[row]) {
            float ap = fmaxf(OPT_P - val, 0.f);
            sp_p += __expf(-S_GAMMA * ap * (val - DELTA_P));
          } else {
            float an = fmaxf(val + MARG, 0.f);
            sn_p += __expf(S_GAMMA * an * (val - MARG));
          }
        }
      }
    }
  }

  double sn_d = (double)sn_p, sp_d = (double)sp_p;
  for (int off = 16; off > 0; off >>= 1) {
    sn_d += __shfl_down(sn_d, off, 32);
    sp_d += __shfl_down(sp_d, off, 32);
  }
  if (lane == 0) {
    atomicAdd(&accum[0], sn_d);
    atomicAdd(&accum[1], sp_d);
  }
}

__global__ void finalize_kernel(const double* __restrict__ accum,
                                float* __restrict__ out) {
  out[0] = (float)log(1.0 + accum[0] * accum[1]);
}

extern "C" void kernel_launch(void* const* d_in, const int* in_sizes, int n_in,
                              void* d_out, int out_size, void* d_ws, size_t ws_size,
                              hipStream_t stream) {
  (void)n_in; (void)out_size; (void)ws_size;
  const int*   labels = (const int*)d_in[1];
  const float* emb    = (const float*)d_in[2];
  const float* W      = (const float*)d_in[3];

  const int B = in_sizes[1];          // 256
  const int D = in_sizes[2] / B;      // 512
  const int C = in_sizes[3] / D;      // 100000

  double* accum = (double*)d_ws;                        // 2 doubles
  bf16_t* ffb   = (bf16_t*)((char*)d_ws + 256);         // B*D bf16

  init_accum_kernel<<<1, 1, 0, stream>>>(accum);
  normalize_rows_kernel<<<B, 256, 0, stream>>>(emb, ffb, D);

  const int nblocks = (C + NCOLS - 1) / NCOLS;          // 782
  circle_fused_kernel<<<nblocks, THREADS, 0, stream>>>(ffb, W, labels, accum, D, C);

  finalize_kernel<<<1, 1, 0, stream>>>(accum, (float*)d_out);
}
